// Model_77429670412314
// MI455X (gfx1250) — matrix operand order repaired and checked by reading, not compiled, not run on hardware
//
#include <hip/hip_runtime.h>
#include <hip/hip_bf16.h>

// MoE top-2 of 8 experts, N=8192 tokens, D=512, F=2048.
// Strategy: route tokens (4x FLOP saving vs dense), bf16 WMMA 16x16x32 with
// f32 accumulation (8x FLOP/instr vs the f32 16x16x4 path), weights + x
// converted once per call into L2-resident bf16 layouts. Token gather uses
// CDNA5 async global->LDS copies (ASYNCcnt) when available.

#define N_TOK 8192
#define DIM   512
#define FDIM  2048
#define NEXP  8
#define EPSV  2.220446049250313e-16f

typedef __attribute__((ext_vector_type(16))) __bf16 v16bf;
typedef __attribute__((ext_vector_type(8)))  float  v8f;
typedef __attribute__((ext_vector_type(4)))  int    v4i;

union ABf { uint4 q[2]; v16bf v; };

#if defined(__AMDGCN__) && __has_builtin(__builtin_amdgcn_global_load_async_to_lds_b128)
#define USE_ASYNC_GATHER 1
typedef __attribute__((address_space(1))) v4i as1_v4i;   // global int4
typedef __attribute__((address_space(3))) v4i as3_v4i;   // LDS int4
#else
#define USE_ASYNC_GATHER 0
#endif

__device__ __forceinline__ unsigned short f2bf(float f) {
    unsigned u = __float_as_uint(f);
    u += 0x7fffu + ((u >> 16) & 1u);          // round-to-nearest-even
    return (unsigned short)(u >> 16);
}

__device__ __forceinline__ v8f wmma_bf16(const v16bf& a, const v16bf& b, const v8f& c) {
    return __builtin_amdgcn_wmma_f32_16x16x32_bf16(
        /*neg_a=*/false, a, /*neg_b=*/false, b,
        /*c_mod=*/(short)0, c, /*reuse_a=*/false, /*reuse_b=*/false);
}

// ---------------------------------------------------------------- utilities
__global__ void zero_counts_kernel(unsigned int* counts) {
    if (threadIdx.x < NEXP) counts[threadIdx.x] = 0u;
}

// plain f32 -> bf16 conversion (x)
__global__ void cvt_bf16_kernel(const float* __restrict__ src,
                                unsigned short* __restrict__ dst, size_t total) {
    size_t idx = (size_t)blockIdx.x * blockDim.x + threadIdx.x;
    if (idx < total) dst[idx] = f2bf(src[idx]);
}

// dst[e][c][r] (bf16) = src[e][r][c] (f32); idx enumerates (e,c,r), r fastest.
__global__ void transpose_cvt_kernel(const float* __restrict__ src,
                                     unsigned short* __restrict__ dst,
                                     int R, int C) {
    size_t idx = (size_t)blockIdx.x * blockDim.x + threadIdx.x;
    size_t total = (size_t)NEXP * R * C;
    if (idx >= total) return;
    int r = (int)(idx % (size_t)R);
    size_t t = idx / (size_t)R;
    int c = (int)(t % (size_t)C);
    int e = (int)(t / (size_t)C);
    dst[idx] = f2bf(src[((size_t)e * R + r) * C + c]);
}

// ---------------------------------------------------------------- gating
__global__ void gating_kernel(const float* __restrict__ gl,
                              unsigned int* __restrict__ counts,
                              unsigned int* __restrict__ lists,
                              float* __restrict__ gates) {
    int n = blockIdx.x * blockDim.x + threadIdx.x;
    if (n >= N_TOK) return;
    float v0 = -3.402823466e38f, v1 = -3.402823466e38f;
    int i0 = 0, i1 = 0;
#pragma unroll
    for (int j = 0; j < NEXP; ++j) {
        float l = gl[n * NEXP + j];
        if (l > v0)      { v1 = v0; i1 = i0; v0 = l; i0 = j; }
        else if (l > v1) { v1 = l; i1 = j; }
    }
    float t  = expf(v1 - v0);       // softmax over the two selected logits
    float g1 = t / (1.0f + t);
    float g0 = 1.0f - g1;
    gates[2 * n + 0] = g0;
    gates[2 * n + 1] = g1;
    unsigned p0 = atomicAdd(&counts[i0], 1u);
    lists[i0 * N_TOK + p0] = (unsigned)n;                 // slot 0
    unsigned p1 = atomicAdd(&counts[i1], 1u);
    lists[i1 * N_TOK + p1] = (unsigned)n | 0x80000000u;   // slot 1
}

// ---------------------------------------------------------------- expert MLP
#define XS 520   // s_x row stride (ushorts): 520*2 bytes, rows 16B aligned
#define HS 72    // s_h row stride (ushorts)

__global__ __launch_bounds__(256) void expert_kernel(
    const unsigned short* __restrict__ xbf,   // [N][D] bf16
    const unsigned short* __restrict__ w1t,   // [E][F][D] bf16
    const float* __restrict__ b1,             // [E][F]
    const unsigned short* __restrict__ w2t,   // [E][D][F] bf16
    const float* __restrict__ b2,             // [E][D]
    const unsigned int* __restrict__ counts,
    const unsigned int* __restrict__ lists,   // [E][N]
    const float* __restrict__ gates,          // [N][2]
    float* __restrict__ pairbuf)              // [2][N][D]
{
    __shared__ __align__(16) unsigned short s_x[32 * XS]; // 32x512 bf16 tokens
    __shared__ __align__(16) unsigned short s_h[32 * HS]; // 32x64  bf16 hidden
    __shared__ int   s_tok[32];
    __shared__ int   s_slot[32];
    __shared__ float s_gate[32];

    const int e = blockIdx.y;
    const unsigned int cnt = counts[e];
    const int tileStart = blockIdx.x * 32;
    if ((unsigned)tileStart >= cnt) return;
    const int valid = min(32, (int)cnt - tileStart);

    const int tid = threadIdx.x;
    if (tid < 32) {
        int r = tid, tok = 0, slot = 0; float g = 0.f;
        if (r < valid) {
            unsigned ent = lists[e * N_TOK + tileStart + r];
            tok  = (int)(ent & 0x7fffffffu);
            slot = (int)(ent >> 31);
            g    = gates[tok * 2 + slot];
        }
        s_tok[r] = tok; s_slot[r] = slot; s_gate[r] = g;
    }
    __syncthreads();

    // gather 32 token rows of bf16 x into LDS, 16B granules
#if USE_ASYNC_GATHER
    for (int i = tid; i < 32 * (DIM / 8); i += 256) {   // 2048 chunks, 8/thread
        int r = i >> 6, c = (i & 63) * 8;
        const unsigned short* gp = xbf + (size_t)s_tok[r] * DIM + c;
        unsigned short* lp = s_x + r * XS + c;
        __builtin_amdgcn_global_load_async_to_lds_b128(
            (as1_v4i*)gp, (as3_v4i*)lp, /*offset=*/0, /*cpol=*/0);
    }
    __builtin_amdgcn_s_wait_asynccnt(0);
    __syncthreads();
#else
    for (int i = tid; i < 32 * (DIM / 8); i += 256) {
        int r = i >> 6, c = (i & 63) * 8;
        *(uint4*)(s_x + r * XS + c) =
            *(const uint4*)(xbf + (size_t)s_tok[r] * DIM + c);
    }
    __syncthreads();
#endif

    const int lane = tid & 31;
    const int wave = tid >> 5;
    const int col  = lane & 15;
    const int selA = (lane < 16) ? 0 : 8;    // A: K {0..7,16..23} | {8..15,24..31}
    const int selB = (lane < 16) ? 0 : 8;    // B: K {0..7,16..23} | {8..15,24..31}

    const int mt = wave & 1;     // M-tile (rows 0-15 or 16-31) for both GEMMs
    const int nt1 = wave >> 1;   // GEMM1 N-tile (of 4)
    const int ntb = wave >> 1;   // GEMM2 N-tile base (of 32, step 4)

    const v8f vzero = {0.f, 0.f, 0.f, 0.f, 0.f, 0.f, 0.f, 0.f};
    v8f c2[8];
#pragma unroll
    for (int j = 0; j < 8; ++j) c2[j] = vzero;

    const unsigned short* w1e = w1t + (size_t)e * FDIM * DIM;
    const unsigned short* w2e = w2t + (size_t)e * DIM * FDIM;

    for (int fc = 0; fc < FDIM; fc += 64) {
        // ---- GEMM1: H[32,64] = relu(Xg[32,512] @ W1[:,fc:fc+64] + b1) ----
        v8f acc = vzero;
        const int f1 = fc + nt1 * 16 + col;                    // output column f
        const unsigned short* brow = w1e + (size_t)f1 * DIM;   // contiguous in K=d
        const unsigned short* arow = s_x + (mt * 16 + col) * XS;
#pragma unroll 4
        for (int k = 0; k < DIM; k += 32) {
            ABf A, B;
            A.q[0] = *(const uint4*)(arow + k + selA);
            A.q[1] = *(const uint4*)(arow + k + selA + 16);
            B.q[0] = *(const uint4*)(brow + k + selB);
            B.q[1] = *(const uint4*)(brow + k + selB + 16);
            acc = wmma_bf16(A.v, B.v, acc);
        }
        {   // bias + relu, write bf16 H tile to LDS
            float bias = b1[e * FDIM + f1];
            int rbase = mt * 16 + ((lane < 16) ? 0 : 8);
            int cpos  = nt1 * 16 + col;
#pragma unroll
            for (int i = 0; i < 8; ++i) {
                float h = acc[i] + bias;
                h = h > 0.f ? h : 0.f;
                s_h[(rbase + i) * HS + cpos] = f2bf(h);
            }
        }
        __syncthreads();

        // ---- GEMM2: C[32,512] += H[32,64] @ W2[fc:fc+64, :512] ----
        const unsigned short* harow = s_h + (mt * 16 + col) * HS;
#pragma unroll
        for (int k0 = 0; k0 < 64; k0 += 32) {
            ABf A;
            A.q[0] = *(const uint4*)(harow + k0 + selA);
            A.q[1] = *(const uint4*)(harow + k0 + selA + 16);
#pragma unroll
            for (int j = 0; j < 8; ++j) {
                int d0 = (ntb + j * 4) * 16 + col;
                const unsigned short* b2row = w2e + (size_t)d0 * FDIM + fc;
                ABf B;
                B.q[0] = *(const uint4*)(b2row + k0 + selB);
                B.q[1] = *(const uint4*)(b2row + k0 + selB + 16);
                c2[j] = wmma_bf16(A.v, B.v, c2[j]);
            }
        }
        __syncthreads();
    }

    // ---- epilogue: pairbuf[slot][tok][d] = gate * exp(out + b2) ----
    {
        int rbase = mt * 16 + ((lane < 16) ? 0 : 8);
#pragma unroll
        for (int j = 0; j < 8; ++j) {
            int d = (ntb + j * 4) * 16 + col;
            float bb = b2[e * DIM + d];
#pragma unroll
            for (int i = 0; i < 8; ++i) {
                int r = rbase + i;
                if (r < valid) {
                    float v = c2[j][i] + bb;
                    pairbuf[((size_t)s_slot[r] * N_TOK + s_tok[r]) * DIM + d] =
                        s_gate[r] * expf(v);
                }
            }
        }
    }
}

// ---------------------------------------------------------------- combine
__global__ void combine_kernel(const float* __restrict__ pairbuf,
                               float* __restrict__ out) {
    size_t i = (size_t)blockIdx.x * blockDim.x + threadIdx.x;
    if (i >= (size_t)N_TOK * DIM) return;
    float s = pairbuf[i] + pairbuf[(size_t)N_TOK * DIM + i];
    s = (s == 0.0f) ? EPSV : s;
    out[i] = logf(s);
}

// ---------------------------------------------------------------- launcher
extern "C" void kernel_launch(void* const* d_in, const int* in_sizes, int n_in,
                              void* d_out, int out_size, void* d_ws, size_t ws_size,
                              hipStream_t stream) {
    const float* x  = (const float*)d_in[0];
    const float* gl = (const float*)d_in[1];
    const float* w1 = (const float*)d_in[2];
    const float* b1 = (const float*)d_in[3];
    const float* w2 = (const float*)d_in[4];
    const float* b2 = (const float*)d_in[5];
    float* out = (float*)d_out;

    char* ws = (char*)d_ws;
    size_t off = 0;
    auto carve = [&](size_t bytes) -> char* {
        char* p = ws + off;
        off = (off + bytes + 255) & ~(size_t)255;
        return p;
    };
    unsigned short* w1t   = (unsigned short*)carve((size_t)NEXP * FDIM * DIM * 2);
    unsigned short* w2t   = (unsigned short*)carve((size_t)NEXP * DIM * FDIM * 2);
    unsigned short* xbf   = (unsigned short*)carve((size_t)N_TOK * DIM * 2);
    float*          pair  = (float*)carve((size_t)2 * N_TOK * DIM * 4);
    float*          gts   = (float*)carve((size_t)2 * N_TOK * 4);
    unsigned int*   cnts  = (unsigned int*)carve(NEXP * 4);
    unsigned int*   lists = (unsigned int*)carve((size_t)NEXP * N_TOK * 4);

    zero_counts_kernel<<<1, 32, 0, stream>>>(cnts);

    size_t xtot = (size_t)N_TOK * DIM;
    cvt_bf16_kernel<<<(int)((xtot + 255) / 256), 256, 0, stream>>>(x, xbf, xtot);

    size_t wtot = (size_t)NEXP * DIM * FDIM;
    int wblocks = (int)((wtot + 255) / 256);
    transpose_cvt_kernel<<<wblocks, 256, 0, stream>>>(w1, w1t, DIM, FDIM);  // -> [E][F][D]
    transpose_cvt_kernel<<<wblocks, 256, 0, stream>>>(w2, w2t, FDIM, DIM);  // -> [E][D][F]

    gating_kernel<<<N_TOK / 256, 256, 0, stream>>>(gl, cnts, lists, gts);

    dim3 egrid(N_TOK / 32, NEXP);   // over-provisioned; blocks past count exit
    expert_kernel<<<egrid, 256, 0, stream>>>(xbf, w1t, b1, w2t, b2,
                                             cnts, lists, gts, pair);

    size_t ototal = (size_t)N_TOK * DIM;
    combine_kernel<<<(int)((ototal + 255) / 256), 256, 0, stream>>>(pair, out);
}
